// RelMultiheadAttention_77601469104405
// MI455X (gfx1250) — matrix operand order repaired and checked by reading, not compiled, not run on hardware
//
#include <hip/hip_runtime.h>
#include <hip/hip_bf16.h>

#define BB 8
#define SS 1024
#define DD 512
#define HH 8
#define AD 64

typedef __attribute__((ext_vector_type(16))) __bf16 v16bf;
typedef __attribute__((ext_vector_type(8)))  __bf16 v8bf;
typedef __attribute__((ext_vector_type(8)))  float  v8f;
typedef __attribute__((ext_vector_type(4)))  float  v4f;

// ---- WMMA wrapper: D = A(16x32 bf16) * B(32x16 bf16) + C(16x16 f32) ----
__device__ inline v8f wmma_bf16(v16bf a, v16bf b, v8f c) {
  return __builtin_amdgcn_wmma_f32_16x16x32_bf16(
      /*neg_a=*/false, a, /*neg_b=*/false, b,
      /*c_mod=*/(short)0, c, /*reuse_a=*/false, /*reuse_b=*/false);
}

// ---- Raw (unconverted) tile loads: issue all VMEM first, convert later ----
struct RawAf { v4f x0, x1, x2, x3; };   // A 16x32 from f32: k = c0+{0..15} contiguous
struct RawBf { v4f x0, x1, x2, x3; };   // B 32x16 from f32: k = c0+{0..15} contiguous

__device__ inline RawAf ldA_f32_raw(const float* __restrict__ row, int kb, int hi) {
  int c0 = kb + hi * 16;
  RawAf r;
  r.x0 = *(const v4f*)(row + c0);
  r.x1 = *(const v4f*)(row + c0 + 4);
  r.x2 = *(const v4f*)(row + c0 + 8);
  r.x3 = *(const v4f*)(row + c0 + 12);
  return r;
}
__device__ inline v16bf cvtA(RawAf r) {
  v16bf a;
#pragma unroll
  for (int i = 0; i < 4; ++i) {
    a[i]      = (__bf16)r.x0[i];
    a[4 + i]  = (__bf16)r.x1[i];
    a[8 + i]  = (__bf16)r.x2[i];
    a[12 + i] = (__bf16)r.x3[i];
  }
  return a;
}

__device__ inline RawBf ldB_f32_raw(const float* __restrict__ col, int kb, int hi) {
  int c0 = kb + hi * 16;
  RawBf r;
  r.x0 = *(const v4f*)(col + c0);
  r.x1 = *(const v4f*)(col + c0 + 4);
  r.x2 = *(const v4f*)(col + c0 + 8);
  r.x3 = *(const v4f*)(col + c0 + 12);
  return r;
}
__device__ inline v16bf cvtB(RawBf r) {
  v16bf b;
#pragma unroll
  for (int i = 0; i < 4; ++i) {
    b[i]      = (__bf16)r.x0[i];
    b[4 + i]  = (__bf16)r.x1[i];
    b[8 + i]  = (__bf16)r.x2[i];
    b[12 + i] = (__bf16)r.x3[i];
  }
  return b;
}

// bf16 direct loads (already in WMMA element type)
__device__ inline v16bf ldA_bf16(const __bf16* __restrict__ row, int kb, int hi) {
  int c0 = kb + hi * 16;
  v8bf lo8 = *(const v8bf*)(row + c0);
  v8bf hi8 = *(const v8bf*)(row + c0 + 8);
  v16bf a;
#pragma unroll
  for (int i = 0; i < 8; ++i) { a[i] = lo8[i]; a[8 + i] = hi8[i]; }
  return a;
}
__device__ inline v16bf ldB_bf16(const __bf16* __restrict__ col, int kb, int hi) {
  return *(const v16bf*)(col + kb + hi * 16);
}

// ---------------------------------------------------------------------------
// Kernel 1: fused input projections.  P = X @ W^T + b  (M=B*S, N=K=512)
// 4 waves/block on adjacent M-tiles (share B through WGP$).
// pid 0: Q -> Qu = Q+U_bias, Qv = Q+V_bias (bf16, row-major)
// pid 1: K -> Kb ; pid 2: V -> Vt (per-head transposed) ; pid 3: R -> Rb
// ---------------------------------------------------------------------------
__global__ __launch_bounds__(128) void proj_kernel(
    const float* __restrict__ q_in, const float* __restrict__ k_in,
    const float* __restrict__ v_in, const float* __restrict__ r_in,
    const float* __restrict__ Wq, const float* __restrict__ bq,
    const float* __restrict__ Wk, const float* __restrict__ bk,
    const float* __restrict__ Wv, const float* __restrict__ bv,
    const float* __restrict__ Wr, const float* __restrict__ br,
    const float* __restrict__ Ubias, const float* __restrict__ Vbias,
    __bf16* __restrict__ Qu, __bf16* __restrict__ Qv,
    __bf16* __restrict__ Kb, __bf16* __restrict__ Vt,
    __bf16* __restrict__ Rb) {
  int lane = threadIdx.x & 31;
  int wave = threadIdx.x >> 5;
  int hi = lane >> 4, lo = lane & 15;
  int m0 = (blockIdx.x * 4 + wave) * 16;
  int n0 = blockIdx.y * 64;
  int pid = blockIdx.z;

  const float *X, *W, *bias;
  if (pid == 0)      { X = q_in; W = Wq; bias = bq; }
  else if (pid == 1) { X = k_in; W = Wk; bias = bk; }
  else if (pid == 2) { X = v_in; W = Wv; bias = bv; }
  else               { X = r_in; W = Wr; bias = br; }

  const float* arow = X + (size_t)(m0 + lo) * DD;
  const float* wrow0 = W + (size_t)(n0 + 0 * 16 + lo) * DD;
  const float* wrow1 = W + (size_t)(n0 + 1 * 16 + lo) * DD;
  const float* wrow2 = W + (size_t)(n0 + 2 * 16 + lo) * DD;
  const float* wrow3 = W + (size_t)(n0 + 3 * 16 + lo) * DD;

  v8f acc[4] = {};
  for (int kb = 0; kb < DD; kb += 32) {
    // Batch all VMEM for this k-step, then convert, then 4 WMMAs.
    RawAf ra  = ldA_f32_raw(arow, kb, hi);
    RawBf rb0 = ldB_f32_raw(wrow0, kb, hi);
    RawBf rb1 = ldB_f32_raw(wrow1, kb, hi);
    RawBf rb2 = ldB_f32_raw(wrow2, kb, hi);
    RawBf rb3 = ldB_f32_raw(wrow3, kb, hi);
    if (kb + 32 < DD) __builtin_prefetch(arow + kb + 32, 0, 1);

    v16bf a = cvtA(ra);
    acc[0] = wmma_bf16(a, cvtB(rb0), acc[0]);
    acc[1] = wmma_bf16(a, cvtB(rb1), acc[1]);
    acc[2] = wmma_bf16(a, cvtB(rb2), acc[2]);
    acc[3] = wmma_bf16(a, cvtB(rb3), acc[3]);
  }

#pragma unroll
  for (int t = 0; t < 4; ++t) {
    int n = n0 + t * 16 + lo;
    float bn = bias[n];
    int h = n / AD, ad = n % AD;
    float ub = Ubias[h * AD + ad], vb = Vbias[h * AD + ad];
#pragma unroll
    for (int r = 0; r < 8; ++r) {
      int m = m0 + hi * 8 + r;
      float cv = acc[t][r] + bn;
      if (pid == 0) {
        Qu[(size_t)m * DD + n] = (__bf16)(cv + ub);
        Qv[(size_t)m * DD + n] = (__bf16)(cv + vb);
      } else if (pid == 1) {
        Kb[(size_t)m * DD + n] = (__bf16)cv;
      } else if (pid == 2) {
        int bi = m / SS, s = m % SS;
        Vt[(((size_t)bi * HH + h) * AD + ad) * SS + s] = (__bf16)cv;
      } else {
        Rb[(size_t)m * DD + n] = (__bf16)cv;
      }
    }
  }
}

// ---------------------------------------------------------------------------
// Kernel 2: AC[bh,q,k] = (Q+U)·K   (which=0, into d_out attn region)
//           BD[bh,q,k] = (Q+V)·R   (which=1, into ws scratch)
// K = AD = 64 -> fully unrolled: 10 vector loads up front, 8 WMMAs.
// ---------------------------------------------------------------------------
__global__ __launch_bounds__(128) void scores_kernel(
    const __bf16* __restrict__ Qu, const __bf16* __restrict__ Qv,
    const __bf16* __restrict__ Kb, const __bf16* __restrict__ Rb,
    float* __restrict__ attn, float* __restrict__ BD) {
  int lane = threadIdx.x & 31;
  int wave = threadIdx.x >> 5;
  int hi = lane >> 4, lo = lane & 15;
  int q0 = (blockIdx.x * 4 + wave) * 16;
  int k0 = blockIdx.y * 64;
  int gz = blockIdx.z;
  int bh = gz & (BB * HH - 1);
  int which = gz >> 6;
  int b = bh / HH, h = bh % HH;

  const __bf16* A  = which ? Qv : Qu;
  const __bf16* Bm = which ? Rb : Kb;
  const __bf16* arow = A + ((size_t)(b * SS) + q0 + lo) * DD + h * AD;

  // Issue all loads for the whole K=64 reduction first.
  v16bf a0 = ldA_bf16(arow, 0, hi);
  v16bf a1 = ldA_bf16(arow, 32, hi);
  v16bf b0[4], b1[4];
#pragma unroll
  for (int t = 0; t < 4; ++t) {
    const __bf16* brow = Bm + ((size_t)(b * SS) + k0 + t * 16 + lo) * DD + h * AD;
    b0[t] = ldB_bf16(brow, 0, hi);
    b1[t] = ldB_bf16(brow, 32, hi);
  }

  v8f acc[4] = {};
#pragma unroll
  for (int t = 0; t < 4; ++t) acc[t] = wmma_bf16(a0, b0[t], acc[t]);
#pragma unroll
  for (int t = 0; t < 4; ++t) acc[t] = wmma_bf16(a1, b1[t], acc[t]);

  float* dst = which ? BD : attn;
#pragma unroll
  for (int t = 0; t < 4; ++t) {
    int k = k0 + t * 16 + lo;
#pragma unroll
    for (int r = 0; r < 8; ++r) {
      int qq = q0 + hi * 8 + r;
      dst[((size_t)bh * SS + qq) * SS + k] = acc[t][r];
    }
  }
}

// ---------------------------------------------------------------------------
// Kernel 3: fused rel_shift + scale + key-padding mask + softmax, in place.
// rel_shift row q == [ BD[q][S-1-q .. S-1], 0, BD[q+1][0 .. S-q-3] ]
// ---------------------------------------------------------------------------
__global__ __launch_bounds__(256) void softmax_kernel(
    float* __restrict__ attn, const float* __restrict__ BD,
    const unsigned char* __restrict__ mask) {
  __shared__ float red[256];
  int q = blockIdx.x;
  int bh = blockIdx.y;
  int b = bh / HH;
  int tid = threadIdx.x;

  float* row = attn + ((size_t)bh * SS + q) * SS;
  const float* bdq  = BD + ((size_t)bh * SS + q) * SS;
  const float* bdq1 = bdq + SS;  // only dereferenced when k > q+1 (so q+1 < S)
  const unsigned char* mrow = mask + (size_t)b * SS;

  float sv[4];
  float mx = -3.0e38f;
#pragma unroll
  for (int i = 0; i < 4; ++i) {
    int k = tid + i * 256;
    float bd;
    if (k <= q)          bd = bdq[SS - 1 - q + k];
    else if (k == q + 1) bd = 0.0f;
    else                 bd = bdq1[k - q - 2];
    float s = (row[k] + bd) * 0.125f;  // 1/sqrt(64)
    if (mrow[k]) s += -1.0e9f;
    sv[i] = s;
    mx = fmaxf(mx, s);
  }

  red[tid] = mx;
  __syncthreads();
  for (int off = 128; off > 0; off >>= 1) {
    if (tid < off) red[tid] = fmaxf(red[tid], red[tid + off]);
    __syncthreads();
  }
  mx = red[0];
  __syncthreads();

  float sum = 0.0f;
#pragma unroll
  for (int i = 0; i < 4; ++i) {
    float e = __expf(sv[i] - mx);
    sv[i] = e;
    sum += e;
  }
  red[tid] = sum;
  __syncthreads();
  for (int off = 128; off > 0; off >>= 1) {
    if (tid < off) red[tid] += red[tid + off];
    __syncthreads();
  }
  float inv = 1.0f / red[0];

#pragma unroll
  for (int i = 0; i < 4; ++i) {
    int k = tid + i * 256;
    row[k] = sv[i] * inv;
  }
}

// ---------------------------------------------------------------------------
// Kernel 4: Hc[b,q,h*64+ad] = sum_k attn[bh,q,k] * V[b,k,h*64+ad]
// A = attn (f32 -> bf16 on load), B = Vt (columns contiguous along k).
// ---------------------------------------------------------------------------
__global__ __launch_bounds__(128) void av_kernel(
    const float* __restrict__ attn, const __bf16* __restrict__ Vt,
    __bf16* __restrict__ Hc) {
  int lane = threadIdx.x & 31;
  int wave = threadIdx.x >> 5;
  int hi = lane >> 4, lo = lane & 15;
  int q0 = (blockIdx.x * 4 + wave) * 16;
  int bh = blockIdx.y;
  int b = bh / HH, h = bh % HH;

  const float* arow = attn + ((size_t)bh * SS + q0 + lo) * SS;
  const __bf16* col0 = Vt + ((size_t)bh * AD + 0 * 16 + lo) * SS;
  const __bf16* col1 = Vt + ((size_t)bh * AD + 1 * 16 + lo) * SS;
  const __bf16* col2 = Vt + ((size_t)bh * AD + 2 * 16 + lo) * SS;
  const __bf16* col3 = Vt + ((size_t)bh * AD + 3 * 16 + lo) * SS;

  v8f acc[4] = {};
  for (int kb = 0; kb < SS; kb += 32) {
    RawAf ra = ldA_f32_raw(arow, kb, hi);
    v16bf bt0 = ldB_bf16(col0, kb, hi);
    v16bf bt1 = ldB_bf16(col1, kb, hi);
    v16bf bt2 = ldB_bf16(col2, kb, hi);
    v16bf bt3 = ldB_bf16(col3, kb, hi);
    if (kb + 32 < SS) __builtin_prefetch(arow + kb + 32, 0, 1);

    v16bf a = cvtA(ra);
    acc[0] = wmma_bf16(a, bt0, acc[0]);
    acc[1] = wmma_bf16(a, bt1, acc[1]);
    acc[2] = wmma_bf16(a, bt2, acc[2]);
    acc[3] = wmma_bf16(a, bt3, acc[3]);
  }

#pragma unroll
  for (int t = 0; t < 4; ++t) {
    int ad = t * 16 + lo;
#pragma unroll
    for (int r = 0; r < 8; ++r) {
      int qq = q0 + hi * 8 + r;
      Hc[((size_t)(b * SS) + qq) * DD + h * AD + ad] = (__bf16)acc[t][r];
    }
  }
}

// ---------------------------------------------------------------------------
// Kernel 5: out = Hc @ Wo^T + bo  (f32 output into d_out[0 .. B*S*D))
// ---------------------------------------------------------------------------
__global__ __launch_bounds__(128) void outproj_kernel(
    const __bf16* __restrict__ Hc, const float* __restrict__ Wo,
    const float* __restrict__ bo, float* __restrict__ out) {
  int lane = threadIdx.x & 31;
  int wave = threadIdx.x >> 5;
  int hi = lane >> 4, lo = lane & 15;
  int m0 = (blockIdx.x * 4 + wave) * 16;
  int n0 = blockIdx.y * 64;

  const __bf16* arow = Hc + (size_t)(m0 + lo) * DD;
  const float* wrow0 = Wo + (size_t)(n0 + 0 * 16 + lo) * DD;
  const float* wrow1 = Wo + (size_t)(n0 + 1 * 16 + lo) * DD;
  const float* wrow2 = Wo + (size_t)(n0 + 2 * 16 + lo) * DD;
  const float* wrow3 = Wo + (size_t)(n0 + 3 * 16 + lo) * DD;

  v8f acc[4] = {};
  for (int kb = 0; kb < DD; kb += 32) {
    v16bf a = ldA_bf16(arow, kb, hi);
    RawBf rb0 = ldB_f32_raw(wrow0, kb, hi);
    RawBf rb1 = ldB_f32_raw(wrow1, kb, hi);
    RawBf rb2 = ldB_f32_raw(wrow2, kb, hi);
    RawBf rb3 = ldB_f32_raw(wrow3, kb, hi);
    if (kb + 32 < DD) __builtin_prefetch(arow + kb + 32, 0, 1);

    acc[0] = wmma_bf16(a, cvtB(rb0), acc[0]);
    acc[1] = wmma_bf16(a, cvtB(rb1), acc[1]);
    acc[2] = wmma_bf16(a, cvtB(rb2), acc[2]);
    acc[3] = wmma_bf16(a, cvtB(rb3), acc[3]);
  }

#pragma unroll
  for (int t = 0; t < 4; ++t) {
    int n = n0 + t * 16 + lo;
    float bn = bo[n];
#pragma unroll
    for (int r = 0; r < 8; ++r) {
      int m = m0 + hi * 8 + r;
      out[(size_t)m * DD + n] = acc[t][r] + bn;
    }
  }
}

// ---------------------------------------------------------------------------
extern "C" void kernel_launch(void* const* d_in, const int* in_sizes, int n_in,
                              void* d_out, int out_size, void* d_ws, size_t ws_size,
                              hipStream_t stream) {
  (void)in_sizes; (void)n_in; (void)out_size; (void)ws_size;

  const float* query  = (const float*)d_in[0];
  const float* key    = (const float*)d_in[1];
  const float* value  = (const float*)d_in[2];
  const float* relpos = (const float*)d_in[3];
  const unsigned char* mask = (const unsigned char*)d_in[4];
  const float* Wq = (const float*)d_in[5];   const float* bq = (const float*)d_in[6];
  const float* Wk = (const float*)d_in[7];   const float* bk = (const float*)d_in[8];
  const float* Wv = (const float*)d_in[9];   const float* bv = (const float*)d_in[10];
  const float* Wr = (const float*)d_in[11];  const float* br = (const float*)d_in[12];
  const float* Wo = (const float*)d_in[13];  const float* bo = (const float*)d_in[14];
  const float* Ub = (const float*)d_in[15];
  const float* Vb = (const float*)d_in[16];

  const size_t BSD = (size_t)BB * SS * DD;  // 4,194,304 elements

  // Workspace layout (needs ~304 MB):
  //  [0..6*BSD) bf16: Qu, Qv, Kb, Rb, Vt, Hc ; then f32 BD[B*H*S*S]
  __bf16* wsb = (__bf16*)d_ws;
  __bf16* Qu = wsb + 0 * BSD;
  __bf16* Qv = wsb + 1 * BSD;
  __bf16* Kb = wsb + 2 * BSD;
  __bf16* Rb = wsb + 3 * BSD;
  __bf16* Vt = wsb + 4 * BSD;
  __bf16* Hc = wsb + 5 * BSD;
  float*  BD = (float*)(wsb + 6 * BSD);

  float* out  = (float*)d_out;
  float* attn = out + BSD;  // (B*H, S, S) attention weights output region

  dim3 g1(BB * SS / 64, DD / 64, 4);        // 128 x 8 x 4, 4 waves/block
  proj_kernel<<<g1, 128, 0, stream>>>(query, key, value, relpos,
                                      Wq, bq, Wk, bk, Wv, bv, Wr, br,
                                      Ub, Vb, Qu, Qv, Kb, Vt, Rb);

  dim3 g2(SS / 64, SS / 64, 2 * BB * HH);   // 16 x 16 x 128
  scores_kernel<<<g2, 128, 0, stream>>>(Qu, Qv, Kb, Rb, attn, BD);

  dim3 g3(SS, BB * HH);                     // 1024 x 64
  softmax_kernel<<<g3, 256, 0, stream>>>(attn, BD, mask);

  dim3 g4(SS / 64, BB * HH);                // 16 x 64
  av_kernel<<<g4, 128, 0, stream>>>(attn, Vt, Hc);

  dim3 g5(BB * SS / 64, DD / 64, 1);        // 128 x 8
  outproj_kernel<<<g5, 128, 0, stream>>>(Hc, Wo, bo, out);
}
